// Self_Attention_New_48404281426203
// MI455X (gfx1250) — matrix operand order repaired and checked by reading, not compiled, not run on hardware
//
#include <hip/hip_runtime.h>
#include <hip/hip_bf16.h>

typedef __attribute__((ext_vector_type(16))) _Float16 v16h;
typedef __attribute__((ext_vector_type(8)))  float    v8f;

union Frag {
    v16h h;
    uint4 u[2];
};

// ---------------------------------------------------------------------------
// f32 -> f16 convert
// ---------------------------------------------------------------------------
__global__ void k_cvt(const float* __restrict__ s, _Float16* __restrict__ d, size_t n) {
    size_t i = (size_t)blockIdx.x * blockDim.x + threadIdx.x;
    if (i < n) d[i] = (_Float16)s[i];
}

// ---------------------------------------------------------------------------
// Tiled WMMA GEMM:  C[M,N] = alpha * (A @ B) + bias[n] + resid
//   AT=false: A stored [M,K] row-major (lda = K-stride)
//   AT=true : A stored [K,M] row-major (lda = M-row length)   (A logically transposed)
//   BT=false: B stored [K,N] row-major
//   BT=true : B stored [N,K] row-major (i.e. computes A @ B^T)
// Block tile 64x256, 8 waves (2x4 wave grid, 2x4 16x16 frags per wave), K-step 32.
// LDS double-buffered with a register relay: next tile's global loads issue
// before the current tile's WMMAs, LDS stores happen after (one barrier/step).
// ---------------------------------------------------------------------------
#define BM 64
#define BN 256
#define BK 32

template <bool AT, bool BT, bool F16OUT>
__global__ void __launch_bounds__(256)
k_gemm(const _Float16* __restrict__ A, const _Float16* __restrict__ Bmat,
       void* __restrict__ Cout, const float* __restrict__ bias,
       const float* __restrict__ resid,
       int M, int N, int K, int lda, int ldb, int ldc,
       long long sA, long long sB, long long sC, long long sR, float alpha)
{
    (void)M; (void)N;
    __shared__ alignas(16) _Float16 As[2][BM][40];   // +8 halves pad, double buffered
    __shared__ alignas(16) _Float16 Bs[2][BN][40];

    const int tid  = threadIdx.x;
    const int lane = tid & 31;
    const int wid  = tid >> 5;
    const int wm   = wid >> 2;     // 0..1  (32 rows each)
    const int wn   = wid & 3;      // 0..3  (64 cols each)
    const int bz   = blockIdx.z;
    const int bm   = blockIdx.y * BM;
    const int bn   = blockIdx.x * BN;

    A    += (size_t)bz * (size_t)sA;
    Bmat += (size_t)bz * (size_t)sB;

    v8f acc[2][4];
    const v8f vzero = {0.f, 0.f, 0.f, 0.f, 0.f, 0.f, 0.f, 0.f};
#pragma unroll
    for (int i = 0; i < 2; ++i)
#pragma unroll
        for (int j = 0; j < 4; ++j) acc[i][j] = vzero;

    // ---- register-relay loaders / LDS stores ----
    auto loadA = [&](int kk) -> uint4 {
        if (!AT) {
            int r = tid >> 2, seg = tid & 3;                       // 64 rows x 4 x 16B
            const _Float16* src = A + (size_t)(bm + r) * lda + kk + seg * 8;
            __builtin_prefetch(src + BK, 0, 1);                    // global_prefetch_b8
            return *reinterpret_cast<const uint4*>(src);
        } else {
            int k = tid >> 3, seg = tid & 7;                       // 32 k-rows x 8 x 16B
            const _Float16* src = A + (size_t)(kk + k) * lda + bm + seg * 8;
            __builtin_prefetch(src + (size_t)BK * lda, 0, 1);
            return *reinterpret_cast<const uint4*>(src);
        }
    };
    auto storeA = [&](int buf, uint4 v) {
        if (!AT) {
            int r = tid >> 2, seg = tid & 3;
            *reinterpret_cast<uint4*>(&As[buf][r][seg * 8]) = v;
        } else {
            int k = tid >> 3, seg = tid & 7;
            const _Float16* hp = reinterpret_cast<const _Float16*>(&v);
#pragma unroll
            for (int j = 0; j < 8; ++j) As[buf][seg * 8 + j][k] = hp[j];
        }
    };
    auto loadB = [&](int kk, int it) -> uint4 {
        int idx = tid + it * 256;                                  // 256 x 32 halves
        if (BT) {
            int n = idx >> 2, seg = idx & 3;
            const _Float16* src = Bmat + (size_t)(bn + n) * ldb + kk + seg * 8;
            __builtin_prefetch(src + BK, 0, 1);
            return *reinterpret_cast<const uint4*>(src);
        } else {
            int k = idx >> 5, seg = idx & 31;
            const _Float16* src = Bmat + (size_t)(kk + k) * ldb + bn + seg * 8;
            __builtin_prefetch(src + (size_t)BK * ldb, 0, 1);
            return *reinterpret_cast<const uint4*>(src);
        }
    };
    auto storeB = [&](int buf, int it, uint4 v) {
        int idx = tid + it * 256;
        if (BT) {
            int n = idx >> 2, seg = idx & 3;
            *reinterpret_cast<uint4*>(&Bs[buf][n][seg * 8]) = v;
        } else {
            int k = idx >> 5, seg = idx & 31;
            const _Float16* hp = reinterpret_cast<const _Float16*>(&v);
#pragma unroll
            for (int j = 0; j < 8; ++j) Bs[buf][seg * 8 + j][k] = hp[j];
        }
    };

    // ---- prologue: stage tile 0 ----
    {
        uint4 ta = loadA(0);
        uint4 tb[4];
#pragma unroll
        for (int it = 0; it < 4; ++it) tb[it] = loadB(0, it);
        storeA(0, ta);
#pragma unroll
        for (int it = 0; it < 4; ++it) storeB(0, it, tb[it]);
    }
    __syncthreads();

    int parity = 0;
    for (int kk = 0; kk < K; kk += BK) {
        // issue next tile's global loads first (latency hidden by WMMAs below)
        const bool has_next = (kk + BK < K);
        uint4 ta = {};
        uint4 tb[4] = {};
        if (has_next) {
            ta = loadA(kk + BK);
#pragma unroll
            for (int it = 0; it < 4; ++it) tb[it] = loadB(kk + BK, it);
        }

        // ---- gather fragments (documented CDNA5 16-bit layouts) ----
        Frag a[2], b[4];
        const bool lo = (lane < 16);
#pragma unroll
        for (int mi = 0; mi < 2; ++mi) {
            int r  = wm * 32 + mi * 16 + (lane & 15);
            int c0 = lo ? 0 : 8;       // A: K {0..7,16..23} / {8..15,24..31}
            a[mi].u[0] = *reinterpret_cast<const uint4*>(&As[parity][r][c0]);
            a[mi].u[1] = *reinterpret_cast<const uint4*>(&As[parity][r][c0 + 16]);
        }
#pragma unroll
        for (int ni = 0; ni < 4; ++ni) {
            int r  = wn * 64 + ni * 16 + (lane & 15);
            int c0 = lo ? 0 : 8;       // B: K {0..7,16..23} / {8..15,24..31}
            b[ni].u[0] = *reinterpret_cast<const uint4*>(&Bs[parity][r][c0]);
            b[ni].u[1] = *reinterpret_cast<const uint4*>(&Bs[parity][r][c0 + 16]);
        }

#pragma unroll
        for (int mi = 0; mi < 2; ++mi)
#pragma unroll
            for (int ni = 0; ni < 4; ++ni)
                acc[mi][ni] = __builtin_amdgcn_wmma_f32_16x16x32_f16(
                    false, a[mi].h, false, b[ni].h,
                    (short)0, acc[mi][ni], false, false);

        // drain the relay into the other LDS buffer
        if (has_next) {
            storeA(parity ^ 1, ta);
#pragma unroll
            for (int it = 0; it < 4; ++it) storeB(parity ^ 1, it, tb[it]);
        }
        __syncthreads();
        parity ^= 1;
    }

    // ---- epilogue: scale, bias, residual, store (C layout: VGPR r -> M = r + 8*(lane>=16)) ----
#pragma unroll
    for (int mi = 0; mi < 2; ++mi) {
#pragma unroll
        for (int ni = 0; ni < 4; ++ni) {
            int row0 = bm + wm * 32 + mi * 16 + ((lane >= 16) ? 8 : 0);
            int col  = bn + wn * 64 + ni * 16 + (lane & 15);
            float badd = bias ? bias[col] : 0.0f;
#pragma unroll
            for (int r = 0; r < 8; ++r) {
                int row = row0 + r;
                float v = acc[mi][ni][r] * alpha + badd;
                if (resid)
                    v += resid[(size_t)bz * (size_t)sR + (size_t)row * ldc + col];
                size_t idx = (size_t)bz * (size_t)sC + (size_t)row * ldc + col;
                if (F16OUT)
                    reinterpret_cast<_Float16*>(Cout)[idx] = (_Float16)v;
                else
                    reinterpret_cast<float*>(Cout)[idx] = v;
            }
        }
    }
}

// ---------------------------------------------------------------------------
// Row softmax (block per row), fp32 in -> f16 out
// ---------------------------------------------------------------------------
__global__ void __launch_bounds__(256)
k_softmax(const float* __restrict__ src, _Float16* __restrict__ dst, int L) {
    const float* x = src + (size_t)blockIdx.x * L;
    _Float16*    y = dst + (size_t)blockIdx.x * L;
    __shared__ float red[256];
    int tid = threadIdx.x;

    float m = -3.4e38f;
    for (int i = tid; i < L; i += 256) m = fmaxf(m, x[i]);
    red[tid] = m; __syncthreads();
    for (int t = 128; t > 0; t >>= 1) {
        if (tid < t) red[tid] = fmaxf(red[tid], red[tid + t]);
        __syncthreads();
    }
    m = red[0]; __syncthreads();

    float s = 0.f;
    for (int i = tid; i < L; i += 256) s += __expf(x[i] - m);
    red[tid] = s; __syncthreads();
    for (int t = 128; t > 0; t >>= 1) {
        if (tid < t) red[tid] += red[tid + t];
        __syncthreads();
    }
    float inv = 1.f / red[0];
    for (int i = tid; i < L; i += 256) y[i] = (_Float16)(__expf(x[i] - m) * inv);
}

// ---------------------------------------------------------------------------
// LayerNorm over [S,H] jointly per batch: stats then apply
// ---------------------------------------------------------------------------
__global__ void __launch_bounds__(1024)
k_ln_stats(const float* __restrict__ x, float* __restrict__ stats, int n) {
    int b = blockIdx.x;
    const float* p = x + (size_t)b * n;
    float s = 0.f, s2 = 0.f;
    for (int i = threadIdx.x; i < n; i += 1024) {
        float v = p[i]; s += v; s2 += v * v;
    }
    __shared__ float rs[1024];
    __shared__ float rq[1024];
    rs[threadIdx.x] = s; rq[threadIdx.x] = s2; __syncthreads();
    for (int t = 512; t > 0; t >>= 1) {
        if (threadIdx.x < t) {
            rs[threadIdx.x] += rs[threadIdx.x + t];
            rq[threadIdx.x] += rq[threadIdx.x + t];
        }
        __syncthreads();
    }
    if (threadIdx.x == 0) {
        float mean = rs[0] / n;
        float var  = rq[0] / n - mean * mean;
        stats[2 * b]     = mean;
        stats[2 * b + 1] = rsqrtf(var + 1e-5f);
    }
}

template <bool F16OUT>
__global__ void __launch_bounds__(256)
k_ln_apply(const float* __restrict__ x, const float* __restrict__ stats,
           void* __restrict__ out, int n) {
    int b = blockIdx.y;
    size_t i = (size_t)blockIdx.x * 256 + threadIdx.x;
    if (i >= (size_t)n) return;
    float mean = stats[2 * b], rstd = stats[2 * b + 1];
    size_t idx = (size_t)b * n + i;
    float v = (x[idx] - mean) * rstd;
    if (F16OUT) reinterpret_cast<_Float16*>(out)[idx] = (_Float16)v;
    else        reinterpret_cast<float*>(out)[idx] = v;
}

// ---------------------------------------------------------------------------
// SGSA gate: gate[row] = dot(x[row,:], w) + b   (wave32 per row)
// ---------------------------------------------------------------------------
__global__ void __launch_bounds__(256)
k_gate(const float* __restrict__ x, const float* __restrict__ w,
       const float* __restrict__ bsg, float* __restrict__ gate, int rows) {
    int wid = threadIdx.x >> 5, lane = threadIdx.x & 31;
    int row = blockIdx.x * 8 + wid;
    if (row >= rows) return;
    const float* p = x + (size_t)row * 768;
    float s = 0.f;
    for (int h = lane; h < 768; h += 32) s += p[h] * w[h];
    for (int off = 16; off > 0; off >>= 1) s += __shfl_xor(s, off, 32);
    if (lane == 0) gate[row] = s + bsg[0];
}

// ---------------------------------------------------------------------------
// pooled[b, 0:H]  = mean_s(gate * x), pooled[b, H:2H] = max_s(gate * x)
// ---------------------------------------------------------------------------
__global__ void __launch_bounds__(256)
k_pool(const float* __restrict__ x, const float* __restrict__ gate,
       float* __restrict__ pooled) {
    const int S_ = 1024, H_ = 768;
    int b = blockIdx.x, hb = blockIdx.y * 128;
    int col = threadIdx.x & 127, sg = threadIdx.x >> 7;
    float sum = 0.f, mx = -3.4e38f;
    for (int s = sg; s < S_; s += 2) {
        float g = gate[b * S_ + s];
        float v = g * x[((size_t)b * S_ + s) * H_ + hb + col];
        sum += v; mx = fmaxf(mx, v);
    }
    __shared__ float r1[256];
    __shared__ float r2[256];
    r1[threadIdx.x] = sum; r2[threadIdx.x] = mx; __syncthreads();
    if (threadIdx.x < 128) {
        float ts = r1[threadIdx.x] + r1[threadIdx.x + 128];
        float tm = fmaxf(r2[threadIdx.x], r2[threadIdx.x + 128]);
        pooled[(size_t)b * 1536 + hb + threadIdx.x]       = ts / (float)S_;
        pooled[(size_t)b * 1536 + 768 + hb + threadIdx.x] = tm;
    }
}

// ---------------------------------------------------------------------------
// Classifier: out[b,c] = pooled[b,:] . W_fnn[c,:] + b_fnn[c]  (C=3, 2H=1536)
// ---------------------------------------------------------------------------
__global__ void __launch_bounds__(256)
k_final(const float* __restrict__ pooled, const float* __restrict__ Wf,
        const float* __restrict__ bf, float* __restrict__ out) {
    int b = blockIdx.x;
    const float* p = pooled + (size_t)b * 1536;
    float a0 = 0.f, a1 = 0.f, a2 = 0.f;
    for (int j = threadIdx.x; j < 1536; j += 256) {
        float v = p[j];
        a0 += v * Wf[j];
        a1 += v * Wf[1536 + j];
        a2 += v * Wf[3072 + j];
    }
    __shared__ float r0[256];
    __shared__ float r1[256];
    __shared__ float r2[256];
    r0[threadIdx.x] = a0; r1[threadIdx.x] = a1; r2[threadIdx.x] = a2;
    __syncthreads();
    for (int t = 128; t > 0; t >>= 1) {
        if (threadIdx.x < t) {
            r0[threadIdx.x] += r0[threadIdx.x + t];
            r1[threadIdx.x] += r1[threadIdx.x + t];
            r2[threadIdx.x] += r2[threadIdx.x + t];
        }
        __syncthreads();
    }
    if (threadIdx.x == 0) {
        out[b * 3 + 0] = r0[0] + bf[0];
        out[b * 3 + 1] = r1[0] + bf[1];
        out[b * 3 + 2] = r2[0] + bf[2];
    }
}

// ---------------------------------------------------------------------------
extern "C" void kernel_launch(void* const* d_in, const int* in_sizes, int n_in,
                              void* d_out, int out_size, void* d_ws, size_t ws_size,
                              hipStream_t stream) {
    (void)in_sizes; (void)n_in; (void)out_size; (void)ws_size;
    const float* tokens = (const float*)d_in[0];
    const float* Wq  = (const float*)d_in[1];
    const float* bq  = (const float*)d_in[2];
    const float* Wk  = (const float*)d_in[3];
    const float* bk  = (const float*)d_in[4];
    const float* Wv  = (const float*)d_in[5];
    const float* bv  = (const float*)d_in[6];
    const float* Wsg = (const float*)d_in[7];
    const float* bsg = (const float*)d_in[8];
    const float* Wf  = (const float*)d_in[9];
    const float* bf  = (const float*)d_in[10];
    float* out = (float*)d_out;

    const int Bb = 32, S = 1024, Hd = 768;
    const size_t HH  = (size_t)Hd * Hd;
    const size_t BSH = (size_t)Bb * S * Hd;
    const size_t BSS = (size_t)Bb * S * S;
    const float SCALE = 0.036084391824351615f;   // 1/sqrt(768)

    char* base = (char*)d_ws;
    size_t off = 0;
    auto take = [&](size_t bytes) -> char* {
        char* p = base + off;
        off += (bytes + 255) & ~(size_t)255;
        return p;
    };
    _Float16* w16   = (_Float16*)take(3 * HH * 2);   // Wq16 | Wk16 | Wv16
    _Float16* tok16 = (_Float16*)take(BSH * 2);
    _Float16* q16   = (_Float16*)take(BSH * 2);      // later Qn16
    _Float16* k16   = (_Float16*)take(BSH * 2);      // later Kn16
    _Float16* v16   = (_Float16*)take(BSH * 2);      // later Vn16
    float*    sc32  = (float*)take(BSS * 4);         // scores, later channel scores [B,H,H]
    _Float16* p16   = (_Float16*)take(BSS * 2);      // probs f16, later channel probs
    float*    fbuf  = (float*)take(BSH * 4);         // attn out / out_n / out_ln2 (in-place)
    _Float16* ln16  = (_Float16*)take(BSH * 2);      // out_ln in f16
    float*    stats = (float*)take(256);
    float*    gate  = (float*)take((size_t)Bb * S * 4);
    float*    pooled= (float*)take((size_t)Bb * 2 * Hd * 4);

    dim3 blk(256);
    const long long sBSH = (long long)S * Hd;
    const long long sBSS = (long long)S * S;
    const long long sBHH = (long long)HH;

    // f32 -> f16 operands
    k_cvt<<<(unsigned)((BSH + 255) / 256), blk, 0, stream>>>(tokens, tok16, BSH);
    k_cvt<<<(unsigned)((HH + 255) / 256), blk, 0, stream>>>(Wq, w16, HH);
    k_cvt<<<(unsigned)((HH + 255) / 256), blk, 0, stream>>>(Wk, w16 + HH, HH);
    k_cvt<<<(unsigned)((HH + 255) / 256), blk, 0, stream>>>(Wv, w16 + 2 * HH, HH);

    // --- token attention ---
    dim3 gProj(Hd / BN, S / BM, Bb);     // (3,16,32)
    k_gemm<false, true, true><<<gProj, blk, 0, stream>>>(tok16, w16,          q16, bq, nullptr,
        S, Hd, Hd, Hd, Hd, Hd, sBSH, 0, sBSH, 0, 1.0f);
    k_gemm<false, true, true><<<gProj, blk, 0, stream>>>(tok16, w16 + HH,     k16, bk, nullptr,
        S, Hd, Hd, Hd, Hd, Hd, sBSH, 0, sBSH, 0, 1.0f);
    k_gemm<false, true, true><<<gProj, blk, 0, stream>>>(tok16, w16 + 2 * HH, v16, bv, nullptr,
        S, Hd, Hd, Hd, Hd, Hd, sBSH, 0, sBSH, 0, 1.0f);

    dim3 gSc(S / BN, S / BM, Bb);        // (4,16,32)
    k_gemm<false, true, false><<<gSc, blk, 0, stream>>>(q16, k16, sc32, nullptr, nullptr,
        S, S, Hd, Hd, Hd, S, sBSH, sBSH, sBSS, 0, SCALE);
    k_softmax<<<Bb * S, blk, 0, stream>>>(sc32, p16, S);
    k_gemm<false, false, false><<<gProj, blk, 0, stream>>>(p16, v16, fbuf, nullptr, nullptr,
        S, Hd, S, S, Hd, Hd, sBSS, sBSH, sBSH, 0, 1.0f);

    // --- LayerNorm 1 -> f16 ---
    k_ln_stats<<<Bb, 1024, 0, stream>>>(fbuf, stats, S * Hd);
    k_ln_apply<true><<<dim3((S * Hd) / 256, Bb), blk, 0, stream>>>(fbuf, stats, ln16, S * Hd);

    // --- channel attention (same weights) ---
    k_gemm<false, true, true><<<gProj, blk, 0, stream>>>(ln16, w16,          q16, bq, nullptr,
        S, Hd, Hd, Hd, Hd, Hd, sBSH, 0, sBSH, 0, 1.0f);
    k_gemm<false, true, true><<<gProj, blk, 0, stream>>>(ln16, w16 + HH,     k16, bk, nullptr,
        S, Hd, Hd, Hd, Hd, Hd, sBSH, 0, sBSH, 0, 1.0f);
    k_gemm<false, true, true><<<gProj, blk, 0, stream>>>(ln16, w16 + 2 * HH, v16, bv, nullptr,
        S, Hd, Hd, Hd, Hd, Hd, sBSH, 0, sBSH, 0, 1.0f);

    dim3 gCh(Hd / BN, Hd / BM, Bb);      // (3,12,32): scores2[d,e] = sum_s Qn[s,d] Kn[s,e]
    k_gemm<true, false, false><<<gCh, blk, 0, stream>>>(q16, k16, sc32, nullptr, nullptr,
        Hd, Hd, S, Hd, Hd, Hd, sBSH, sBSH, sBHH, 0, SCALE);
    k_softmax<<<Bb * Hd, blk, 0, stream>>>(sc32, p16, Hd);
    // out_n = Vn @ attn_n + tokens (residual in epilogue)
    k_gemm<false, false, false><<<gProj, blk, 0, stream>>>(v16, p16, fbuf, nullptr, tokens,
        S, Hd, Hd, Hd, Hd, Hd, sBSH, sBHH, sBSH, sBSH, 1.0f);

    // --- LayerNorm 2 (in place, fp32) ---
    k_ln_stats<<<Bb, 1024, 0, stream>>>(fbuf, stats, S * Hd);
    k_ln_apply<false><<<dim3((S * Hd) / 256, Bb), blk, 0, stream>>>(fbuf, stats, fbuf, S * Hd);

    // --- SGSA gate, pool, classifier ---
    k_gate<<<(Bb * S) / 8, blk, 0, stream>>>(fbuf, Wsg, bsg, gate, Bb * S);
    k_pool<<<dim3(Bb, Hd / 128), blk, 0, stream>>>(fbuf, gate, pooled);
    k_final<<<Bb, blk, 0, stream>>>(pooled, Wf, bf, out);
}
